// FusedMoE_29042568855938
// MI455X (gfx1250) — matrix operand order repaired and checked by reading, not compiled, not run on hardware
//
#include <hip/hip_runtime.h>

// ---------------------------------------------------------------------------
// Fused top-2 MoE for MI455X (gfx1250, wave32, WMMA bf16 16x16x32)
// T=1024 tokens, H=1024 hidden, I=2816 intermediate, E=8 experts, top-2.
// ---------------------------------------------------------------------------

#define T_TOK 1024
#define H_DIM 1024
#define I_DIM 2816
#define TWO_I 5632
#define N_EXP 8

#define MT 32            // token rows per block tile
#define NT 128           // output cols per block tile
#define KT 32            // K per WMMA step
#define LSTR 40          // LDS row stride in ushorts (80B, 16B aligned)

typedef __attribute__((ext_vector_type(16))) __bf16          v16bf;
typedef __attribute__((ext_vector_type(2)))  __bf16          v2bf;
typedef __attribute__((ext_vector_type(8)))  float           v8f;
typedef __attribute__((ext_vector_type(8)))  unsigned short  us8;
typedef __attribute__((ext_vector_type(4)))  unsigned short  us4;
typedef __attribute__((ext_vector_type(2)))  unsigned int    u32x2;
typedef __attribute__((ext_vector_type(2)))  int             i32x2;

struct U16x2 { us8 lo; us8 hi; };

#if defined(__has_builtin)
#if __has_builtin(__builtin_amdgcn_global_load_async_to_lds_b64) && \
    __has_builtin(__builtin_amdgcn_s_wait_asynccnt)
#define USE_ASYNC_LDS 1
#endif
#endif

// fp32 -> bf16 via native conversion (backend pairs adjacent converts into
// v_cvt_pk_bf16_f32 when available; otherwise emits RNE software sequence)
__device__ __forceinline__ unsigned short f2bf(float f) {
    __bf16 h = (__bf16)f;
    return __builtin_bit_cast(unsigned short, h);
}

__device__ __forceinline__ unsigned int pk_bf16(float a, float b) {
#if defined(__has_builtin) && __has_builtin(__builtin_amdgcn_cvt_pk_bf16_f32)
    v2bf v = __builtin_amdgcn_cvt_pk_bf16_f32(a, b);
    return __builtin_bit_cast(unsigned int, v);
#else
    v2bf v;
    v.x = (__bf16)a;
    v.y = (__bf16)b;
    return __builtin_bit_cast(unsigned int, v);
#endif
}

__device__ __forceinline__ float silu(float x) {
    return x / (1.0f + __expf(-x));
}

// A fragment: lane L -> row m=L&15 (+wave row base), K blocks {kb, 16+kb}, kb=8*(L>>4)
__device__ __forceinline__ v16bf ld_fragA(const unsigned short* row, int kb) {
    U16x2 f;
    f.lo = *(const us8*)(row + kb);
    f.hi = *(const us8*)(row + 16 + kb);
    return __builtin_bit_cast(v16bf, f);
}
// B fragment: lane L -> col n=L&15 (+base), K blocks {kb, 16+kb}, kb=8*(L>>4); LDS is [n][k]
__device__ __forceinline__ v16bf ld_fragB(const unsigned short* p) {
    U16x2 f;
    f.lo = *(const us8*)(p);
    f.hi = *(const us8*)(p + 16);
    return __builtin_bit_cast(v16bf, f);
}

// ---------------------------------------------------------------------------
// 1) Router: top-2 of 8 logits, renormalized softmax weights
// ---------------------------------------------------------------------------
__global__ __launch_bounds__(256) void moe_router(
    const float* __restrict__ logits, int* __restrict__ eid, float* __restrict__ wts)
{
    int t = blockIdx.x * 256 + threadIdx.x;
    if (t >= T_TOK) return;
    float l[N_EXP];
#pragma unroll
    for (int e = 0; e < N_EXP; ++e) l[e] = logits[t * N_EXP + e];
    int i0 = 0; float m0 = l[0];
#pragma unroll
    for (int e = 1; e < N_EXP; ++e) if (l[e] > m0) { m0 = l[e]; i0 = e; }
    int i1 = -1; float m1 = -3.0e38f;
#pragma unroll
    for (int e = 0; e < N_EXP; ++e) if (e != i0 && l[e] > m1) { m1 = l[e]; i1 = e; }
    float r  = __expf(m1 - m0);
    float w0 = 1.0f / (1.0f + r);
    eid[2 * t] = i0;  eid[2 * t + 1] = i1;
    wts[2 * t] = w0;  wts[2 * t + 1] = 1.0f - w0;
}

// ---------------------------------------------------------------------------
// 2) Deterministic per-expert compaction: wave w handles expert w (wave32 ballot)
// ---------------------------------------------------------------------------
__global__ __launch_bounds__(256) void moe_build(
    const int* __restrict__ eid, int* __restrict__ list,
    int* __restrict__ pos, int* __restrict__ offs)
{
    __shared__ int cnts[N_EXP];
    int tid = threadIdx.x;
    int w = tid >> 5, lane = tid & 31;
    int cnt = 0;
    for (int base = 0; base < T_TOK; base += 32) {
        int t = base + lane;
        bool m = (eid[2 * t] == w) || (eid[2 * t + 1] == w);
        unsigned msk = __builtin_amdgcn_ballot_w32(m);
        cnt += __popc(msk);
    }
    if (lane == 0) cnts[w] = cnt;
    __syncthreads();
    int off = 0;
    for (int i = 0; i < w; ++i) off += cnts[i];
    if (tid == 0) {
        int s = 0;
        for (int i = 0; i < N_EXP; ++i) { offs[i] = s; s += cnts[i]; }
        offs[N_EXP] = s;
    }
    for (int base = 0; base < T_TOK; base += 32) {
        int t = base + lane;
        bool m = (eid[2 * t] == w) || (eid[2 * t + 1] == w);
        unsigned msk = __builtin_amdgcn_ballot_w32(m);
        if (m) {
            int s = off + __popc(msk & ((1u << lane) - 1u));
            list[s] = t;
            int k = (eid[2 * t] == w) ? 0 : 1;
            pos[2 * t + k] = s;
        }
        off += __popc(msk);
    }
}

// ---------------------------------------------------------------------------
// 3) GEMM1: act[slot, i] = silu(x . w13_gate) * (x . w13_up), bf16 WMMA
//    block = (i-tile 128, token-tile 32, expert); 8 waves: 2(m) x 4(n)
// ---------------------------------------------------------------------------
__global__ __launch_bounds__(256) void moe_gemm1(
    const float* __restrict__ x, const float* __restrict__ w13,
    const int* __restrict__ list, const int* __restrict__ offs,
    unsigned short* __restrict__ act)
{
    __shared__ __attribute__((aligned(32))) unsigned short sA[MT * LSTR];
    __shared__ __attribute__((aligned(32))) unsigned short sBg[NT * LSTR];
    __shared__ __attribute__((aligned(32))) unsigned short sBu[NT * LSTR];

    const int e   = blockIdx.z;
    const int mt  = blockIdx.y;
    const int nt  = blockIdx.x;
    const int o0  = offs[e];
    const int cnt = offs[e + 1] - o0;
    if (mt * MT >= cnt) return;

    const int tid = threadIdx.x, wid = tid >> 5, lane = tid & 31;
    const int wm = wid & 1, wn = wid >> 1;
    const int c0 = nt * NT;

    v8f aG0 = {}, aG1 = {}, aU0 = {}, aU1 = {};

    const int ar = tid >> 3;           // A row 0..31
    const int ac = (tid & 7) * 4;      // A col seg
    int tokRow = -1;
    if (mt * MT + ar < cnt) tokRow = list[o0 + mt * MT + ar];

    const int bn  = tid & 127;         // B col 0..127 (one n per thread)
    const int bkh = (tid >> 7) * 16;   // 16-deep k strip: 0 or 16

    // prefetch lane mapping: 256 lines of next weight tile (gate+up)
    const int pfm = tid >> 7;                 // 0=gate,1=up
    const int pfr = (tid & 127) >> 2;         // k row 0..31
    const int pfs = (tid & 3) * 32;           // 128B line segment

    const int m   = (lane & 15) + wm * 16;
    const int kbA = (lane >> 4) * 8;
    const int nl  = lane & 15;
    const int kbB = (lane >> 4) * 8;

    for (int kk = 0; kk < H_DIM; kk += KT) {
        __syncthreads();
        {   // A: gathered token rows fp32 -> bf16 (packed)
            u32x2 v = {};
            if (tokRow >= 0) {
                const float4 p = *(const float4*)(x + (size_t)tokRow * H_DIM + kk + ac);
                v.x = pk_bf16(p.x, p.y);
                v.y = pk_bf16(p.z, p.w);
            }
            *(u32x2*)&sA[ar * LSTR + ac] = v;
        }
        {   // B: w13 gate + up columns, [n][k] LDS, packed dword stores
            const size_t base = (size_t)e * H_DIM * TWO_I
                              + (size_t)(kk + bkh) * TWO_I + (c0 + bn);
#pragma unroll
            for (int kp = 0; kp < 16; kp += 2) {
                float g0 = w13[base + (size_t)kp * TWO_I];
                float g1 = w13[base + (size_t)(kp + 1) * TWO_I];
                float u0 = w13[base + (size_t)kp * TWO_I + I_DIM];
                float u1 = w13[base + (size_t)(kp + 1) * TWO_I + I_DIM];
                *(unsigned int*)&sBg[bn * LSTR + bkh + kp] = pk_bf16(g0, g1);
                *(unsigned int*)&sBu[bn * LSTR + bkh + kp] = pk_bf16(u0, u1);
            }
        }
        // prefetch next K-step's weight tile (one lane per 128B cacheline)
        if (kk + KT < H_DIM) {
            const float* pf = w13 + (size_t)e * H_DIM * TWO_I
                            + (size_t)(kk + KT + pfr) * TWO_I
                            + (c0 + pfs + pfm * I_DIM);
            __builtin_prefetch(pf, 0, 1);
        }
        __syncthreads();

        v16bf a  = ld_fragA(&sA[m * LSTR], kbA);
        v16bf g0 = ld_fragB(&sBg[(wn * 32 + nl) * LSTR + kbB]);
        v16bf g1 = ld_fragB(&sBg[(wn * 32 + 16 + nl) * LSTR + kbB]);
        v16bf u0 = ld_fragB(&sBu[(wn * 32 + nl) * LSTR + kbB]);
        v16bf u1 = ld_fragB(&sBu[(wn * 32 + 16 + nl) * LSTR + kbB]);

        aG0 = __builtin_amdgcn_wmma_f32_16x16x32_bf16(false, a, false, g0, (short)0, aG0, false, false);
        aG1 = __builtin_amdgcn_wmma_f32_16x16x32_bf16(false, a, false, g1, (short)0, aG1, false, false);
        aU0 = __builtin_amdgcn_wmma_f32_16x16x32_bf16(false, a, false, u0, (short)0, aU0, false, false);
        aU1 = __builtin_amdgcn_wmma_f32_16x16x32_bf16(false, a, false, u1, (short)0, aU1, false, false);
    }

    // SwiGLU epilogue -> bf16 act rows (compact slot layout)
    const int rsel = lane >> 4;
#pragma unroll
    for (int j = 0; j < 8; ++j) {
        int rloc = wm * 16 + j + 8 * rsel;
        int r = mt * MT + rloc;
        if (r < cnt) {
            size_t row = (size_t)(o0 + r) * I_DIM;
            int col = c0 + wn * 32 + nl;
            act[row + col]      = f2bf(silu(aG0[j]) * aU0[j]);
            act[row + col + 16] = f2bf(silu(aG1[j]) * aU1[j]);
        }
    }
}

// ---------------------------------------------------------------------------
// 4) GEMM2: partial[slot, h] = act[slot, :] . w2[e, :, h], bf16 WMMA
// ---------------------------------------------------------------------------
__global__ __launch_bounds__(256) void moe_gemm2(
    const unsigned short* __restrict__ act, const float* __restrict__ w2,
    const int* __restrict__ offs, float* __restrict__ partial)
{
    __shared__ __attribute__((aligned(32))) unsigned short sA[MT * LSTR];
    __shared__ __attribute__((aligned(32))) unsigned short sB[NT * LSTR];

    const int e   = blockIdx.z;
    const int mt  = blockIdx.y;
    const int nt  = blockIdx.x;
    const int o0  = offs[e];
    const int cnt = offs[e + 1] - o0;
    if (mt * MT >= cnt) return;

    const int tid = threadIdx.x, wid = tid >> 5, lane = tid & 31;
    const int wm = wid & 1, wn = wid >> 1;
    const int c0 = nt * NT;

    v8f a0 = {}, a1 = {};

    const int ar = tid >> 3;
    const int ac = (tid & 7) * 4;
    const int arow = mt * MT + ar;
    const bool avalid = arow < cnt;

    const int bn  = tid & 127;
    const int bkh = (tid >> 7) * 16;

    const int pfr = (tid & 127) >> 2;
    const int pfs = (tid & 3) * 32;

    const int m   = (lane & 15) + wm * 16;
    const int kbA = (lane >> 4) * 8;
    const int nl  = lane & 15;
    const int kbB = (lane >> 4) * 8;

    for (int kk = 0; kk < I_DIM; kk += KT) {
        __syncthreads();
        {   // A: act already bf16 -> pure copy into LDS
#if defined(USE_ASYNC_LDS)
            if (avalid)
                __builtin_amdgcn_global_load_async_to_lds_b64(
                    (i32x2*)(act + (size_t)(o0 + arow) * I_DIM + kk + ac),
                    (i32x2*)&sA[ar * LSTR + ac],
                    0, 0);
#else
            // stale data in invalid rows only feeds masked output rows
            if (avalid) {
                us4 v = *(const us4*)(act + (size_t)(o0 + arow) * I_DIM + kk + ac);
                *(us4*)&sA[ar * LSTR + ac] = v;
            }
#endif
        }
        {   // B: w2[e][k][h], [n][k] LDS, packed dword stores
            const size_t base = ((size_t)e * I_DIM + kk + bkh) * H_DIM + (c0 + bn);
#pragma unroll
            for (int kp = 0; kp < 16; kp += 2) {
                float b0 = w2[base + (size_t)kp * H_DIM];
                float b1 = w2[base + (size_t)(kp + 1) * H_DIM];
                *(unsigned int*)&sB[bn * LSTR + bkh + kp] = pk_bf16(b0, b1);
            }
        }
        if (kk + KT < I_DIM && tid < 128) {
            const float* pf = w2 + ((size_t)e * I_DIM + kk + KT + pfr) * H_DIM + (c0 + pfs);
            __builtin_prefetch(pf, 0, 1);
        }
#if defined(USE_ASYNC_LDS)
        __builtin_amdgcn_s_wait_asynccnt(0);
#endif
        __syncthreads();

        v16bf a  = ld_fragA(&sA[m * LSTR], kbA);
        v16bf b0 = ld_fragB(&sB[(wn * 32 + nl) * LSTR + kbB]);
        v16bf b1 = ld_fragB(&sB[(wn * 32 + 16 + nl) * LSTR + kbB]);

        a0 = __builtin_amdgcn_wmma_f32_16x16x32_bf16(false, a, false, b0, (short)0, a0, false, false);
        a1 = __builtin_amdgcn_wmma_f32_16x16x32_bf16(false, a, false, b1, (short)0, a1, false, false);
    }

    const int rsel = lane >> 4;
#pragma unroll
    for (int j = 0; j < 8; ++j) {
        int rloc = wm * 16 + j + 8 * rsel;
        int r = mt * MT + rloc;
        if (r < cnt) {
            size_t row = (size_t)(o0 + r) * H_DIM;
            int col = c0 + wn * 32 + nl;
            partial[row + col]      = a0[j];
            partial[row + col + 16] = a1[j];
        }
    }
}

// ---------------------------------------------------------------------------
// 5) Deterministic weighted combine: out[t] = w0*partial[pos0] + w1*partial[pos1]
// ---------------------------------------------------------------------------
__global__ __launch_bounds__(256) void moe_combine(
    const float* __restrict__ partial, const float* __restrict__ wts,
    const int* __restrict__ pos, float* __restrict__ out)
{
    int t = blockIdx.x;
    int h = threadIdx.x * 4;
    float w0 = wts[2 * t], w1 = wts[2 * t + 1];
    int p0 = pos[2 * t], p1 = pos[2 * t + 1];
    float4 a = *(const float4*)(partial + (size_t)p0 * H_DIM + h);
    float4 b = *(const float4*)(partial + (size_t)p1 * H_DIM + h);
    float4 r;
    r.x = w0 * a.x + w1 * b.x;
    r.y = w0 * a.y + w1 * b.y;
    r.z = w0 * a.z + w1 * b.z;
    r.w = w0 * a.w + w1 * b.w;
    *(float4*)(out + (size_t)t * H_DIM + h) = r;
}

// ---------------------------------------------------------------------------
extern "C" void kernel_launch(void* const* d_in, const int* in_sizes, int n_in,
                              void* d_out, int out_size, void* d_ws, size_t ws_size,
                              hipStream_t stream)
{
    const float* hidden = (const float*)d_in[0];   // [1024, 1024]
    const float* rlog   = (const float*)d_in[1];   // [1024, 8]
    const float* w13    = (const float*)d_in[2];   // [8, 1024, 5632]
    const float* w2     = (const float*)d_in[3];   // [8, 2816, 1024]
    float* out = (float*)d_out;                    // [1024, 1024]

    char* ws = (char*)d_ws;
    int*   eid  = (int*)(ws);                         //  8 KB
    float* wts  = (float*)(ws + 8192);                //  8 KB
    int*   pos  = (int*)(ws + 16384);                 //  8 KB
    int*   list = (int*)(ws + 24576);                 //  8 KB
    int*   offs = (int*)(ws + 32768);                 //  9 ints (pad to 256B)
    unsigned short* act = (unsigned short*)(ws + 33024);              // 2048*2816*2 = 11.5 MB
    float* partial = (float*)(ws + 33024 + (size_t)2048 * I_DIM * 2); // 2048*1024*4 = 8 MB

    moe_router <<<dim3(4),                      256, 0, stream>>>(rlog, eid, wts);
    moe_build  <<<dim3(1),                      256, 0, stream>>>(eid, list, pos, offs);
    moe_gemm1  <<<dim3(I_DIM / NT, 32, N_EXP),  256, 0, stream>>>(hidden, w13, list, offs, act);
    moe_gemm2  <<<dim3(H_DIM / NT, 32, N_EXP),  256, 0, stream>>>(act, w2, offs, partial);
    moe_combine<<<dim3(T_TOK),                  256, 0, stream>>>(partial, wts, pos, out);
}
